// SBN_75986561401043
// MI455X (gfx1250) — matrix operand order repaired and checked by reading, not compiled, not run on hardware
//
#include <hip/hip_runtime.h>
#include <hip/hip_bf16.h>
#include <math.h>

#define NDIGIT  10
#define DIM_IN  256
#define DIM_OUT 784
#define BATCH   16384

typedef __attribute__((ext_vector_type(16))) __bf16 v16bf;
typedef __attribute__((ext_vector_type(8)))  float  v8f;

// Pack two f32 into one dword of two bf16 (truncation) with a single v_perm_b32.
__device__ __forceinline__ unsigned int pack_bf16(float lo, float hi) {
    return __builtin_amdgcn_perm(__builtin_bit_cast(unsigned int, hi),
                                 __builtin_bit_cast(unsigned int, lo),
                                 0x07060302u);
}

// ---------------- binning kernels ----------------

__global__ void SBN_zero(int* counts, int* cursor) {
    int t = threadIdx.x;
    if (t < NDIGIT) { counts[t] = 0; cursor[t] = 0; }
}

__global__ void SBN_count(const int* __restrict__ label, int* counts) {
    int b = blockIdx.x * blockDim.x + threadIdx.x;
    if (b < BATCH) atomicAdd(&counts[label[b]], 1);
}

__global__ void SBN_prefix(const int* __restrict__ counts, int* offsets) {
    if (threadIdx.x == 0 && blockIdx.x == 0) {
        int acc = 0;
        for (int i = 0; i < NDIGIT; ++i) { offsets[i] = acc; acc += counts[i]; }
    }
}

__global__ void SBN_scatter(const int* __restrict__ label,
                            const int* __restrict__ offsets,
                            int* cursor, int* __restrict__ perm) {
    int b = blockIdx.x * blockDim.x + threadIdx.x;
    if (b < BATCH) {
        int l = label[b];
        int p = atomicAdd(&cursor[l], 1);
        perm[offsets[l] + p] = b;
    }
}

// ---- WMMA GEMM: per label, out = sigmoid(theta_rows * W[l]^T + b[l]) ----
// NT = number of 16-wide column subtiles handled per wave (compile-time).
// All columns touched by an instantiation are guaranteed in-bounds by the launch.

template <int NT>
__global__ __launch_bounds__(128)
void SBN_moe_gemm(const float* __restrict__ theta,
                  const float* __restrict__ megaW,
                  const float* __restrict__ megab,
                  const int*   __restrict__ perm,
                  const int*   __restrict__ counts,
                  const int*   __restrict__ offsets,
                  float*       __restrict__ out,
                  int colOffset) {
    const int lane = threadIdx.x & 31;
    const int wave = threadIdx.x >> 5;
    const int l    = blockIdx.z;
    const int cnt  = counts[l];

    const int rowtile = blockIdx.y * 4 + wave;
    const int rowbase = rowtile * 16;
    if (rowbase >= cnt) return;                 // wave-uniform exit
    const int valid = min(16, cnt - rowbase);
    const int start = offsets[l];
    const int colbase = colOffset + blockIdx.x * (NT * 16);

    const int m  = lane & 15;                   // row (A) / col (B) within 16-tile
    const int kh = lane >> 4;                   // K-half select

    // A-matrix source row (clamped for partial tiles; stores are guarded later)
    const int mc   = (m < valid) ? m : (valid - 1);
    const int arow = perm[start + rowbase + mc];
    const float* __restrict__ Ap = theta + (size_t)arow * DIM_IN;

    // B-matrix source rows: W[l][col][:]  (out[m,n] = sum_k A[m,k]*W[n,k])
    const float* __restrict__ Bp[NT];
#pragma unroll
    for (int nt = 0; nt < NT; ++nt) {
        int col = colbase + nt * 16 + m;
        Bp[nt] = megaW + ((size_t)l * DIM_OUT + col) * DIM_IN;
    }

    v8f acc[NT];
#pragma unroll
    for (int nt = 0; nt < NT; ++nt)
#pragma unroll
        for (int j = 0; j < 8; ++j) acc[nt][j] = 0.0f;

#pragma unroll
    for (int kb = 0; kb < DIM_IN / 32; ++kb) {
        const int k0 = kb * 32;

        if (kb + 1 < DIM_IN / 32) {
            __builtin_prefetch(Ap + k0 + 32 + kh * 8, 0, 0);
        }

        // ---- A fragment (16x32 bf16): lanes 0-15: K 0..15; lanes 16-31: K 16..31
        union { v16bf v; unsigned int d[8]; } afr;
        {
            const float* s0 = Ap + k0 + kh * 16;
            const float* s1 = Ap + k0 + 8 + kh * 16;
#pragma unroll
            for (int j = 0; j < 4; ++j) afr.d[j]     = pack_bf16(s0[2*j], s0[2*j + 1]);
#pragma unroll
            for (int j = 0; j < 4; ++j) afr.d[4 + j] = pack_bf16(s1[2*j], s1[2*j + 1]);
        }

        // ---- B fragments (32x16 bf16): lanes 0-15 hold K 0..15, lanes 16-31 hold K 16..31
#pragma unroll
        for (int nt = 0; nt < NT; ++nt) {
            union { v16bf v; unsigned int d[8]; } bfr;
            const float* t = Bp[nt] + k0 + kh * 16;
#pragma unroll
            for (int j = 0; j < 8; ++j) bfr.d[j] = pack_bf16(t[2*j], t[2*j + 1]);
            acc[nt] = __builtin_amdgcn_wmma_f32_16x16x32_bf16(
                /*neg_a=*/false, afr.v, /*neg_b=*/false, bfr.v,
                /*c_mod=*/(short)0, acc[nt], /*reuse_a=*/false, /*reuse_b=*/false);
        }
    }

    // ---- epilogue: bias + sigmoid + scatter back by original batch index ----
    // C layout: VGPR i -> M = i (lanes 0-15) or 8+i (lanes 16-31); N = lane & 15
    const int mrow0 = kh * 8;
    int orow[8];
#pragma unroll
    for (int i = 0; i < 8; ++i) {
        int mm = mrow0 + i;
        orow[i] = (mm < valid) ? perm[start + rowbase + mm] : -1;
    }

#pragma unroll
    for (int nt = 0; nt < NT; ++nt) {
        int col = colbase + nt * 16 + m;
        float bias = megab[l * DIM_OUT + col];
#pragma unroll
        for (int i = 0; i < 8; ++i) {
            if (orow[i] >= 0) {
                float x = acc[nt][i] + bias;
                out[(size_t)orow[i] * DIM_OUT + col] = 1.0f / (1.0f + __expf(-x));
            }
        }
    }
}

// ---------------- launcher ----------------

extern "C" void kernel_launch(void* const* d_in, const int* in_sizes, int n_in,
                              void* d_out, int out_size, void* d_ws, size_t ws_size,
                              hipStream_t stream) {
    const float* theta = (const float*)d_in[0];
    const int*   label = (const int*)d_in[1];
    const float* megaW = (const float*)d_in[2];
    const float* megab = (const float*)d_in[3];
    float* out = (float*)d_out;

    int* counts  = (int*)d_ws;       // [0..9]
    int* offsets = counts + 16;      // [16..25]
    int* cursor  = counts + 32;      // [32..41]
    int* perm    = counts + 64;      // [64 .. 64+16384)

    SBN_zero<<<1, 32, 0, stream>>>(counts, cursor);
    SBN_count<<<BATCH / 256, 256, 0, stream>>>(label, counts);
    SBN_prefix<<<1, 32, 0, stream>>>(counts, offsets);
    SBN_scatter<<<BATCH / 256, 256, 0, stream>>>(label, offsets, cursor, perm);

    // 784 = 12*64 + 16: 12 full 64-col tiles (NT=4) + one 16-col tail tile (NT=1).
    // Up to 1024 row tiles / 4 waves per block = 256 groups; 10 labels.
    dim3 gridMain(12, 256, NDIGIT);
    SBN_moe_gemm<4><<<gridMain, 128, 0, stream>>>(theta, megaW, megab, perm,
                                                  counts, offsets, out, /*colOffset=*/0);
    dim3 gridTail(1, 256, NDIGIT);
    SBN_moe_gemm<1><<<gridTail, 128, 0, stream>>>(theta, megaW, megab, perm,
                                                  counts, offsets, out, /*colOffset=*/768);
}
